// AttentionBlock_23862838297409
// MI455X (gfx1250) — matrix operand order repaired and checked by reading, not compiled, not run on hardware
//
#include <hip/hip_runtime.h>
#include <hip/hip_bf16.h>
#include <math.h>

// ---------------- problem constants (match reference) ----------------
#define Bq   4
#define Sq   2048
#define Dq   1024
#define Hq   16
#define HDq  64
#define FFq  4096
#define NROW (Bq * Sq)          // 8192 token rows

// ---------------- WMMA types / helpers ----------------
typedef __attribute__((ext_vector_type(16))) __bf16 v16bf;
typedef __attribute__((ext_vector_type(8)))  float  v8f;

union BF16Frag {
  unsigned int   u32[8];
  unsigned short u16[16];
  v16bf          v;
};

__device__ __forceinline__ unsigned short f2bf(float f) {
  // round-to-nearest-even fp32 -> bf16
  unsigned int u = __float_as_uint(f);
  u += 0x7FFFu + ((u >> 16) & 1u);
  return (unsigned short)(u >> 16);
}

__device__ __forceinline__ v8f wmma_bf16(const v16bf& a, const v16bf& b, const v8f& c) {
  return __builtin_amdgcn_wmma_f32_16x16x32_bf16(false, a, false, b, (short)0, c,
                                                 false, false);
}

// ---------------- CDNA5 async global->LDS staging (ASYNCcnt path) ----------------
__device__ __forceinline__ unsigned lds_off(const void* p) {
  // generic -> LDS address space; AS(3) pointers are workgroup-relative byte offsets
  return (unsigned)(unsigned long long)(__attribute__((address_space(3))) const void*)p;
}

__device__ __forceinline__ void async_load_b32(unsigned ldsoff, const unsigned int* g) {
  // GLOBAL_LOAD_ASYNC_TO_LDS_B32: VDST = per-lane LDS byte address, VADDR = 64-bit global
  asm volatile("global_load_async_to_lds_b32 %0, %1, off"
               :: "v"(ldsoff), "v"(g) : "memory");
}

__device__ __forceinline__ void wait_async_zero() {
  asm volatile("s_wait_asynccnt 0x0" ::: "memory");
}

// ---------------- one-shot fp32 -> bf16 weight conversion ----------------
__global__ __launch_bounds__(256) void cvt_bf16_kernel(const float4* __restrict__ in,
                                                       unsigned int* __restrict__ out,
                                                       int n4) {
  (void)n4;
  const int i = blockIdx.x * 256 + threadIdx.x;   // grid sized exactly to n/4
  const float4 v = in[i];
  out[i * 2 + 0] = (unsigned int)f2bf(v.x) | ((unsigned int)f2bf(v.y) << 16);
  out[i * 2 + 1] = (unsigned int)f2bf(v.z) | ((unsigned int)f2bf(v.w) << 16);
}

// ---------------- LayerNorm: fp32 in, bf16 out; one block per row ----------------
__global__ __launch_bounds__(256) void ln_kernel(const float* __restrict__ x,
                                                 const float* __restrict__ w,
                                                 const float* __restrict__ b,
                                                 unsigned short* __restrict__ out) {
  const int row = blockIdx.x;
  const float* xr = x + (size_t)row * Dq;
  float v[4];
  float s = 0.f;
#pragma unroll
  for (int i = 0; i < 4; i++) { v[i] = xr[threadIdx.x + i * 256]; s += v[i]; }
  __shared__ float red[8];
#pragma unroll
  for (int m = 16; m >= 1; m >>= 1) s += __shfl_xor(s, m, 32);
  const int wave = threadIdx.x >> 5;
  if ((threadIdx.x & 31) == 0) red[wave] = s;
  __syncthreads();
  float mu = 0.f;
#pragma unroll
  for (int i = 0; i < 8; i++) mu += red[i];
  mu *= (1.0f / Dq);
  float vs = 0.f;
#pragma unroll
  for (int i = 0; i < 4; i++) { float d = v[i] - mu; vs += d * d; }
#pragma unroll
  for (int m = 16; m >= 1; m >>= 1) vs += __shfl_xor(vs, m, 32);
  __syncthreads();
  if ((threadIdx.x & 31) == 0) red[wave] = vs;
  __syncthreads();
  float var = 0.f;
#pragma unroll
  for (int i = 0; i < 8; i++) var += red[i];
  var *= (1.0f / Dq);
  const float rstd = rsqrtf(var + 1e-5f);
#pragma unroll
  for (int i = 0; i < 4; i++) {
    const int c = threadIdx.x + i * 256;
    out[(size_t)row * Dq + c] = f2bf((v[i] - mu) * rstd * w[c] + b[c]);
  }
}

// ---------------- bf16 WMMA GEMM, block tile 128x128, wave tile 32x64 ----------------
// A: [M,K] bf16, W: [K,N] bf16.
// mode 0: outF = act?(acc+bias) + resid   (fp32)
// mode 1: outB = bf16(act?(acc+bias))
// mode 2: QKV scatter: col -> (head, hd, which); k->kB, q->qB (scaled 1/8), v->vB
#define TM 128
#define TN 128
#define KS 32

__global__ __launch_bounds__(256) void gemm_bf16_kernel(
    const unsigned short* __restrict__ A, const unsigned short* __restrict__ W,
    const float* __restrict__ bias, const float* __restrict__ resid,
    float* __restrict__ outF, unsigned short* __restrict__ outB,
    unsigned short* __restrict__ qB, unsigned short* __restrict__ kB,
    unsigned short* __restrict__ vB,
    int M, int N, int K, int act, int mode) {
  __shared__ __align__(16) unsigned short lA[TM][KS + 2];
  __shared__ __align__(16) unsigned short lB[TN][KS + 2];

  const int t = threadIdx.x;
  const int lane = t & 31;
  const int wave = t >> 5;
  const int waveM = wave >> 1;   // 0..3 -> 32 rows each
  const int waveN = wave & 1;    // 0..1 -> 64 cols each
  const int sub = lane >> 4;     // half-wave 0/1
  const int l16 = lane & 15;
  const int m0blk = blockIdx.y * TM;
  const int n0blk = blockIdx.x * TN;

  const unsigned int* A32 = (const unsigned int*)A;
  const unsigned int* W32 = (const unsigned int*)W;
  const int K2 = K >> 1, N2 = N >> 1;

  v8f acc[2][4];
#pragma unroll
  for (int mi = 0; mi < 2; mi++)
#pragma unroll
    for (int ni = 0; ni < 4; ni++) acc[mi][ni] = (v8f)0.f;

  for (int k0 = 0; k0 < K; k0 += KS) {
    __syncthreads();
    // stage A tile 128x32 bf16 via async global->LDS (memory-order, no VGPR landing)
#pragma unroll
    for (int i = 0; i < 8; i++) {
      const int e = t + i * 256;
      const int r = e >> 4, c = e & 15;                 // 16 dwords per 32-elem row
      async_load_b32(lds_off(&lA[r][c << 1]),
                     A32 + (size_t)(m0blk + r) * K2 + (k0 >> 1) + c);
    }
    // stage W tile 32x128 bf16, transposed into LDS [n][k] (VGPR path: transpose)
#pragma unroll
    for (int i = 0; i < 8; i++) {
      const int e = t + i * 256;
      const int kk = e >> 6, np = e & 63;               // dword = 2 adjacent n
      const unsigned int w2 = W32[(size_t)(k0 + kk) * N2 + (n0blk >> 1) + np];
      lB[2 * np + 0][kk] = (unsigned short)(w2 & 0xFFFFu);
      lB[2 * np + 1][kk] = (unsigned short)(w2 >> 16);
    }
    wait_async_zero();
    __syncthreads();

    BF16Frag af[2];
#pragma unroll
    for (int mi = 0; mi < 2; mi++) {
      const int m = waveM * 32 + mi * 16 + l16;
#pragma unroll
      for (int j = 0; j < 8; j++) {
        const int kk = ((j >> 2) << 4) + (sub << 3) + ((j & 3) << 1);  // ISA A-layout
        af[mi].u32[j] = *(const unsigned int*)&lA[m][kk];
      }
    }
#pragma unroll
    for (int ni = 0; ni < 4; ni++) {
      BF16Frag bf;
      const int n = waveN * 64 + ni * 16 + l16;
      const unsigned int* p = (const unsigned int*)&lB[n][0];   // ISA B-layout
#pragma unroll
      for (int j = 0; j < 8; j++) bf.u32[j] = p[((j >> 2) << 3) + (sub << 2) + (j & 3)];
#pragma unroll
      for (int mi = 0; mi < 2; mi++)
        acc[mi][ni] = wmma_bf16(af[mi].v, bf.v, acc[mi][ni]);
    }
  }

  // epilogue
#pragma unroll
  for (int mi = 0; mi < 2; mi++)
#pragma unroll
    for (int ni = 0; ni < 4; ni++) {
      const int col = n0blk + waveN * 64 + ni * 16 + l16;
      const float bval = bias ? bias[col] : 0.f;
#pragma unroll
      for (int r = 0; r < 8; r++) {
        const int row = m0blk + waveM * 32 + mi * 16 + sub * 8 + r;
        float vv = acc[mi][ni][r] + bval;
        if (act) vv = 0.5f * vv * (1.f + erff(vv * 0.70710678118654752f));
        if (mode == 0) {
          if (resid) vv += resid[(size_t)row * N + col];
          outF[(size_t)row * N + col] = vv;
        } else if (mode == 1) {
          outB[(size_t)row * N + col] = f2bf(vv);
        } else {
          const int which = col % 3;
          const int head = col / 192;
          const int hd = (col % 192) / 3;
          const size_t o = (size_t)row * Dq + head * HDq + hd;
          if (which == 0)      kB[o] = f2bf(vv);
          else if (which == 1) qB[o] = f2bf(vv * 0.125f);   // fold 1/sqrt(HD)
          else                 vB[o] = f2bf(vv);
        }
      }
    }
}

// ---------------- causal flash attention (bf16 WMMA, fp32 accum) ----------------
// qB/kB/vB: [B*S, D] bf16, col = head*64 + hd (Q pre-scaled). ctx out: bf16 same layout.
__global__ __launch_bounds__(128) void flash_kernel(const unsigned short* __restrict__ qB,
                                                    const unsigned short* __restrict__ kB,
                                                    const unsigned short* __restrict__ vB,
                                                    unsigned short* __restrict__ ctx) {
  const int qt = blockIdx.x;     // 64-row q tile
  const int h  = blockIdx.y;
  const int b  = blockIdx.z;
  const int t = threadIdx.x;
  const int lane = t & 31;
  const int wave = t >> 5;       // 4 waves, 16 q rows each
  const int sub = lane >> 4;
  const int l16 = lane & 15;
  const size_t rowbase = (size_t)b * Sq;
  const int qbase = qt * 64;

  __shared__ __align__(16) unsigned short lK[64][HDq + 2];   // [key][hd]
  __shared__ __align__(16) unsigned short lV[HDq][64 + 2];   // [hd][key]
  __shared__ __align__(16) unsigned short lP[64][64 + 2];    // [qrow][key]

  // Q fragments straight from global bf16 (even-K pairs match ISA A-layout)
  BF16Frag qf[2];
  {
    const int m = qbase + wave * 16 + l16;
    const unsigned int* q32 =
        (const unsigned int*)(qB + (rowbase + m) * (size_t)Dq + h * HDq);
#pragma unroll
    for (int ks = 0; ks < 2; ks++)
#pragma unroll
      for (int j = 0; j < 8; j++) {
        const int kk = ks * 32 + ((j >> 2) << 4) + (sub << 3) + ((j & 3) << 1);
        qf[ks].u32[j] = q32[kk >> 1];
      }
  }

  v8f accc[4];
#pragma unroll
  for (int nt = 0; nt < 4; nt++) accc[nt] = (v8f)0.f;
  float mrow[8], lrow[8];
#pragma unroll
  for (int r = 0; r < 8; r++) { mrow[r] = -INFINITY; lrow[r] = 0.f; }

  const unsigned int* K32 = (const unsigned int*)kB;
  const unsigned int* V32 = (const unsigned int*)vB;

  for (int kb = 0; kb <= qt; kb++) {
    const int kbase = kb * 64;
    __syncthreads();
    // stage K via async global->LDS; V transposed via VGPR path (2048 dwords each)
#pragma unroll
    for (int i = 0; i < 16; i++) {
      const int e = t + i * 128;
      const int key = e >> 5, dp = e & 31;              // dword = 2 adjacent hd
      const size_t base = (rowbase + kbase + key) * (size_t)(Dq >> 1) + h * (HDq >> 1);
      async_load_b32(lds_off(&lK[key][dp << 1]), K32 + base + dp);
      const unsigned int v2 = V32[base + dp];
      lV[2 * dp + 0][key] = (unsigned short)(v2 & 0xFFFFu);
      lV[2 * dp + 1][key] = (unsigned short)(v2 >> 16);
    }
    wait_async_zero();
    __syncthreads();

    // scores = Q @ K^T  (16 q rows x 64 keys per wave)
    v8f sacc[4];
#pragma unroll
    for (int nt = 0; nt < 4; nt++) sacc[nt] = (v8f)0.f;
#pragma unroll
    for (int ks = 0; ks < 2; ks++) {
#pragma unroll
      for (int nt = 0; nt < 4; nt++) {
        BF16Frag bfK;
        const int key = nt * 16 + l16;
        const unsigned int* p = (const unsigned int*)&lK[key][ks * 32];
#pragma unroll
        for (int j = 0; j < 8; j++) bfK.u32[j] = p[((j >> 2) << 3) + (sub << 2) + (j & 3)];
        sacc[nt] = wmma_bf16(qf[ks].v, bfK.v, sacc[nt]);
      }
    }

    // causal mask + online softmax
    const int qr_base = qbase + wave * 16 + sub * 8;
#pragma unroll
    for (int r = 0; r < 8; r++) {
      const int qr = qr_base + r;
      float vals[4];
      float mx = -INFINITY;
#pragma unroll
      for (int nt = 0; nt < 4; nt++) {
        float vv = sacc[nt][r];
        if (kbase + nt * 16 + l16 > qr) vv = -INFINITY;
        vals[nt] = vv;
        mx = fmaxf(mx, vv);
      }
#pragma unroll
      for (int mk = 8; mk >= 1; mk >>= 1) mx = fmaxf(mx, __shfl_xor(mx, mk, 16));
      const float mnew  = fmaxf(mrow[r], mx);
      const float alpha = __expf(mrow[r] - mnew);
      float ssum = 0.f;
#pragma unroll
      for (int nt = 0; nt < 4; nt++) {
        const float p = __expf(vals[nt] - mnew);
        ssum += p;
        lP[wave * 16 + sub * 8 + r][nt * 16 + l16] = f2bf(p);
      }
#pragma unroll
      for (int mk = 8; mk >= 1; mk >>= 1) ssum += __shfl_xor(ssum, mk, 16);
      lrow[r] = lrow[r] * alpha + ssum;
      mrow[r] = mnew;
#pragma unroll
      for (int nt = 0; nt < 4; nt++) accc[nt][r] *= alpha;
    }

    // wave-local LDS RAW: drain DS counter before re-reading our P strip
    asm volatile("s_wait_dscnt 0" ::: "memory");
    __builtin_amdgcn_wave_barrier();

    // ctx += P @ V
#pragma unroll
    for (int ks = 0; ks < 2; ks++) {
      BF16Frag afP;
      const int m = wave * 16 + l16;
#pragma unroll
      for (int j = 0; j < 8; j++) {
        const int kk = ks * 32 + ((j >> 2) << 4) + (sub << 3) + ((j & 3) << 1);
        afP.u32[j] = *(const unsigned int*)&lP[m][kk];
      }
#pragma unroll
      for (int nt = 0; nt < 4; nt++) {
        BF16Frag bfV;
        const int d = nt * 16 + l16;
        const unsigned int* p = (const unsigned int*)&lV[d][ks * 32];
#pragma unroll
        for (int j = 0; j < 8; j++) bfV.u32[j] = p[((j >> 2) << 3) + (sub << 2) + (j & 3)];
        accc[nt] = wmma_bf16(afP.v, bfV.v, accc[nt]);
      }
    }
  }

  // normalize and write ctx bf16 (col = head*64 + hd)
#pragma unroll
  for (int nt = 0; nt < 4; nt++) {
    const int col = h * HDq + nt * 16 + l16;
#pragma unroll
    for (int r = 0; r < 8; r++) {
      const int row = qbase + wave * 16 + sub * 8 + r;
      ctx[(rowbase + row) * (size_t)Dq + col] = f2bf(accc[nt][r] / lrow[r]);
    }
  }
}

// ---------------- host-side orchestration ----------------
extern "C" void kernel_launch(void* const* d_in, const int* in_sizes, int n_in,
                              void* d_out, int out_size, void* d_ws, size_t ws_size,
                              hipStream_t stream) {
  (void)in_sizes; (void)n_in; (void)out_size; (void)ws_size;
  const float* x      = (const float*)d_in[0];
  const float* ln1_w  = (const float*)d_in[1];
  const float* ln1_b  = (const float*)d_in[2];
  const float* w_qkv  = (const float*)d_in[3];
  const float* w_proj = (const float*)d_in[4];
  const float* b_proj = (const float*)d_in[5];
  const float* ln2_w  = (const float*)d_in[6];
  const float* ln2_b  = (const float*)d_in[7];
  const float* w_ff1  = (const float*)d_in[8];
  const float* b_ff1  = (const float*)d_in[9];
  const float* w_ff2  = (const float*)d_in[10];
  const float* b_ff2  = (const float*)d_in[11];
  float* out = (float*)d_out;

  // scratch layout (bytes), lifetime-aliased, 136 MB total:
  //  [0,6M)   wqkv_b   [6M,8M)  wproj_b  [8M,16M) wff1_b  [16M,24M) wff2_b
  //  [24M,40M) hb (ln1 out; later ln2 out)
  //  [40M,56M) Qb  [56M,72M) Kb  [72M,88M) Vb      (dead after flash)
  //  [88M,104M) ctxb                                (dead after proj)
  //  [104M,136M) x1 (fp32)
  //  [40M,104M) ffb  (aliases Q/K/V + ctxb after they die)
  char* ws = (char*)d_ws;
  const size_t MB = 1024 * 1024;
  unsigned short* wqkvb = (unsigned short*)(ws + 0);
  unsigned short* wprojb = (unsigned short*)(ws + 6 * MB);
  unsigned short* wff1b = (unsigned short*)(ws + 8 * MB);
  unsigned short* wff2b = (unsigned short*)(ws + 16 * MB);
  unsigned short* hb    = (unsigned short*)(ws + 24 * MB);
  unsigned short* Qb    = (unsigned short*)(ws + 40 * MB);
  unsigned short* Kb    = (unsigned short*)(ws + 56 * MB);
  unsigned short* Vb    = (unsigned short*)(ws + 72 * MB);
  unsigned short* ctxb  = (unsigned short*)(ws + 88 * MB);
  float*          x1    = (float*)(ws + 104 * MB);
  unsigned short* ffb   = (unsigned short*)(ws + 40 * MB);

  dim3 blk256(256), blk128(128);

  // 0) weights -> bf16 (once per call; deterministic, no caching)
  cvt_bf16_kernel<<<(Dq * 3 * Dq) / 1024, blk256, 0, stream>>>((const float4*)w_qkv,
                                                               (unsigned int*)wqkvb,
                                                               (Dq * 3 * Dq) / 4);
  cvt_bf16_kernel<<<(Dq * Dq) / 1024, blk256, 0, stream>>>((const float4*)w_proj,
                                                           (unsigned int*)wprojb,
                                                           (Dq * Dq) / 4);
  cvt_bf16_kernel<<<(Dq * FFq) / 1024, blk256, 0, stream>>>((const float4*)w_ff1,
                                                            (unsigned int*)wff1b,
                                                            (Dq * FFq) / 4);
  cvt_bf16_kernel<<<(FFq * Dq) / 1024, blk256, 0, stream>>>((const float4*)w_ff2,
                                                            (unsigned int*)wff2b,
                                                            (FFq * Dq) / 4);

  // 1) hb = bf16(LN1(x))
  ln_kernel<<<NROW, blk256, 0, stream>>>(x, ln1_w, ln1_b, hb);
  // 2) qkv = hb @ wqkv, scattered into Qb(scaled)/Kb/Vb
  gemm_bf16_kernel<<<dim3(3 * Dq / TN, NROW / TM), blk256, 0, stream>>>(
      hb, wqkvb, nullptr, nullptr, nullptr, nullptr, Qb, Kb, Vb,
      NROW, 3 * Dq, Dq, 0, 2);
  // 3) ctxb = causal_flash(Qb, Kb, Vb)
  flash_kernel<<<dim3(Sq / 64, Hq, Bq), blk128, 0, stream>>>(Qb, Kb, Vb, ctxb);
  // 4) x1 = x + ctxb @ wproj + b_proj        (fp32)
  gemm_bf16_kernel<<<dim3(Dq / TN, NROW / TM), blk256, 0, stream>>>(
      ctxb, wprojb, b_proj, x, x1, nullptr, nullptr, nullptr, nullptr,
      NROW, Dq, Dq, 0, 0);
  // 5) hb = bf16(LN2(x1))
  ln_kernel<<<NROW, blk256, 0, stream>>>(x1, ln2_w, ln2_b, hb);
  // 6) ffb = bf16(gelu(hb @ wff1 + b_ff1))
  gemm_bf16_kernel<<<dim3(FFq / TN, NROW / TM), blk256, 0, stream>>>(
      hb, wff1b, b_ff1, nullptr, nullptr, ffb, nullptr, nullptr, nullptr,
      NROW, FFq, Dq, 1, 1);
  // 7) out = x1 + ffb @ wff2 + b_ff2          (fp32)
  gemm_bf16_kernel<<<dim3(Dq / TN, NROW / TM), blk256, 0, stream>>>(
      ffb, wff2b, b_ff2, x1, out, nullptr, nullptr, nullptr, nullptr,
      NROW, Dq, FFq, 0, 0);
}
